// QLSTMCell_61512521613689
// MI455X (gfx1250) — matrix operand order repaired and checked by reading, not compiled, not run on hardware
//
#include <hip/hip_runtime.h>

typedef _Float16 v16h __attribute__((ext_vector_type(16)));
typedef float    v8f  __attribute__((ext_vector_type(8)));

union HV { uint4 u[2]; v16h h; };
union PK { _Float16 h[2]; unsigned int u; };
union ST { _Float16 h[8]; uint4 u; };

#define PI_4 0.78539816339744830962f

// ---------------------------------------------------------------------------
// K1: proj = LN(tanh([x,h] @ W_in^T + b)) -> angles -> product state st0 (f16)
// one wave per sample, 8 samples per 256-thread block
// ---------------------------------------------------------------------------
__global__ __launch_bounds__(256) void k_prep(
    const float* __restrict__ x_t, const float* __restrict__ h_prev,
    const float* __restrict__ W_in, const float* __restrict__ b_in,
    const float* __restrict__ ln_g, const float* __restrict__ ln_b,
    _Float16* __restrict__ st0h)
{
    __shared__ float wlds[8 * 512];
    int t = threadIdx.x;
    {   // cooperative load of W_in (8x512 f32 = 16KB)
        const float4* src = (const float4*)W_in;
        float4* dst = (float4*)wlds;
        #pragma unroll
        for (int i = 0; i < 4; ++i) { int idx = i * 256 + t; dst[idx] = src[idx]; }
    }
    __syncthreads();

    int wave = t >> 5, lane = t & 31;
    long s = (long)blockIdx.x * 8 + wave;
    int k0 = lane * 16;
    const float* vsrc = (k0 < 256) ? (x_t + s * 256 + k0)
                                   : (h_prev + s * 256 + (k0 - 256));
    float v[16];
    {
        const float4* p = (const float4*)vsrc;
        #pragma unroll
        for (int i = 0; i < 4; ++i) {
            float4 f = p[i];
            v[i*4+0] = f.x; v[i*4+1] = f.y; v[i*4+2] = f.z; v[i*4+3] = f.w;
        }
    }
    float acc[8];
    #pragma unroll
    for (int q = 0; q < 8; ++q) {
        float a = 0.f;
        const float* wr = wlds + q * 512 + k0;
        #pragma unroll
        for (int j = 0; j < 16; ++j) a = fmaf(wr[j], v[j], a);
        acc[q] = a;
    }
    #pragma unroll
    for (int m = 16; m >= 1; m >>= 1) {
        #pragma unroll
        for (int q = 0; q < 8; ++q) acc[q] += __shfl_xor(acc[q], m, 32);
    }
    float proj[8], mu = 0.f;
    #pragma unroll
    for (int q = 0; q < 8; ++q) { proj[q] = tanhf(acc[q] + b_in[q]); mu += proj[q]; }
    mu *= 0.125f;
    float var = 0.f;
    #pragma unroll
    for (int q = 0; q < 8; ++q) { float d = proj[q] - mu; var += d * d; }
    var *= 0.125f;
    float inv = rsqrtf(var + 1e-5f);
    float u0[8], u1[8];
    #pragma unroll
    for (int q = 0; q < 8; ++q) {
        float th = (proj[q] - mu) * inv * ln_g[q] + ln_b[q];
        float a = 0.5f * th + PI_4;           // H then RY(th): (c-s)/rt2, (c+s)/rt2
        u0[q] = cosf(a);
        u1[q] = sinf(a);
    }
    ST out;
    #pragma unroll
    for (int tt = 0; tt < 8; ++tt) {
        int b = lane * 8 + tt;
        float p = 1.0f;
        #pragma unroll
        for (int q = 0; q < 8; ++q) {
            int bit = (b >> (7 - q)) & 1;
            p *= bit ? u1[q] : u0[q];
        }
        out.h[tt] = (_Float16)p;
    }
    *(uint4*)(st0h + s * 256 + lane * 8) = out.u;
}

// ---------------------------------------------------------------------------
// K2: build U (fixed weight circuit) per VQC by evolving identity columns in
// LDS; repack U^T into WMMA B-fragment order (f16, packed u32 pairs).
// 8 blocks: v = blk>>1, half = blk&1 (128 columns each), 128 threads.
// ---------------------------------------------------------------------------
__global__ __launch_bounds__(128) void k_buildU(
    const float* __restrict__ w_i, const float* __restrict__ w_f,
    const float* __restrict__ w_g, const float* __restrict__ w_o,
    unsigned int* __restrict__ UhB)   // [4][32768] u32
{
    extern __shared__ float S[];      // [256 states][128 cols]
    int t = threadIdx.x;
    int v = blockIdx.x >> 1, half = blockIdx.x & 1;
    int cg = half * 128 + t;          // global column (== k of U^T)
    const float* W = (v == 0) ? w_i : (v == 1) ? w_f : (v == 2) ? w_g : w_o;

    for (int b = 0; b < 256; ++b) S[b * 128 + t] = (b == cg) ? 1.0f : 0.0f;

    const int cn0[7] = {0, 2, 4, 6, 1, 3, 5};
    const int cn1[7] = {1, 3, 5, 7, 2, 4, 6};
    for (int layer = 0; layer < 4; ++layer) {
        for (int g = 0; g < 7; ++g) {
            int mc = 1 << (7 - cn0[g]), mt = 1 << (7 - cn1[g]);
            for (int b = 0; b < 256; ++b) {
                if ((b & mc) && !(b & mt)) {
                    float a0 = S[b * 128 + t], a1 = S[(b | mt) * 128 + t];
                    S[b * 128 + t] = a1;
                    S[(b | mt) * 128 + t] = a0;
                }
            }
        }
        for (int q = 0; q < 8; ++q) {
            float th = W[layer * 8 + q];
            float cc = cosf(0.5f * th), ss = sinf(0.5f * th);
            int m = 1 << (7 - q);
            for (int bl = 0; bl < 128; ++bl) {
                int low = bl & (m - 1);
                int b = ((bl - low) << 1) | low;   // bit q == 0
                float a0 = S[b * 128 + t], a1 = S[(b | m) * 128 + t];
                S[b * 128 + t] = cc * a0 - ss * a1;
                S[(b | m) * 128 + t] = ss * a0 + cc * a1;
            }
        }
    }
    __syncthreads();
    // S[n][c] = U[n][c]  =>  U^T[k][n] = S[n][k]
    // B-frag order: u32 index i: j=i&7, lane=(i>>3)&31, nb=(i>>8)&15, ks=i>>12
    //   k0 = ks*32 + (lane>=16 ? 16 : 0) + 2j ;  n = nb*16 + (lane&15)
    unsigned int* dst = UhB + v * 32768;
    for (int r = 0; r < 128; ++r) {
        int i = half * 16384 + r * 128 + t;
        int j = i & 7, lane = (i >> 3) & 31, nb = (i >> 8) & 15, ks = i >> 12;
        int k0 = ks * 32 + ((lane >> 4) << 4) + 2 * j;
        int n = nb * 16 + (lane & 15);
        int kl = k0 - half * 128;
        PK pk;
        pk.h[0] = (_Float16)S[n * 128 + kl];
        pk.h[1] = (_Float16)S[n * 128 + kl + 1];
        dst[i] = pk.u;
    }
}

// ---------------------------------------------------------------------------
// K3: Y = ST0 @ U^T via v_wmma_f32_16x16x32_f16; epilogue squares + signed
// reduce -> Z[B][4][8]. 8 waves/WG, 128 samples/WG, U staged in 128KB LDS.
// ---------------------------------------------------------------------------
__global__ __launch_bounds__(256) void k_gemm(
    const _Float16* __restrict__ st0h,
    const unsigned int* __restrict__ UhB,
    float* __restrict__ Zall)
{
    extern __shared__ unsigned int Blds[];   // 32768 u32 = 128KB
    int t = threadIdx.x, lane = t & 31, wave = t >> 5;
    int ln = lane & 15;
    long sbase = (long)blockIdx.x * 128 + wave * 16;
    long srow = sbase + ln;

    // per-lane signs for measurement bits 3..0 of n (qubits 4..7)
    const float sg4 = (ln & 8) ? -1.f : 1.f;
    const float sg5 = (ln & 4) ? -1.f : 1.f;
    const float sg6 = (ln & 2) ? -1.f : 1.f;
    const float sg7 = (ln & 1) ? -1.f : 1.f;

    for (int v = 0; v < 4; ++v) {
        __syncthreads();
        {   // stage B-matrix fragments for this VQC
            const uint4* src = (const uint4*)(UhB + v * 32768);
            uint4* dst4 = (uint4*)Blds;
            #pragma unroll
            for (int i = 0; i < 32; ++i) dst4[i * 256 + t] = src[i * 256 + t];
        }
        __syncthreads();

        v8f acc[16];
        #pragma unroll
        for (int nb = 0; nb < 16; ++nb) acc[nb] = (v8f){0.f,0.f,0.f,0.f,0.f,0.f,0.f,0.f};

        for (int ks = 0; ks < 8; ++ks) {
            // A fragment: lanes<16 -> K {0..15}; lanes>=16 -> {16..31}
            int kA = ks * 32 + ((lane >> 4) << 4);
            HV A;
            const uint4* ap = (const uint4*)(st0h + srow * 256 + kA);
            A.u[0] = ap[0];     // K kA..kA+7
            A.u[1] = ap[1];     // K kA+8..kA+15
            #pragma unroll
            for (int nb = 0; nb < 16; ++nb) {
                HV Bf;
                const uint4* bp = (const uint4*)Blds + ((ks * 16 + nb) * 32 + lane) * 2;
                Bf.u[0] = bp[0];
                Bf.u[1] = bp[1];
                acc[nb] = __builtin_amdgcn_wmma_f32_16x16x32_f16(
                    false, A.h, false, Bf.h, (short)0, acc[nb], false, false);
            }
        }

        // epilogue: p = y^2 ; z[i] = sum_n p * (1 - 2*bit_{7-i}(n))
        float z[8][8];
        #pragma unroll
        for (int r = 0; r < 8; ++r)
            #pragma unroll
            for (int i = 0; i < 8; ++i) z[r][i] = 0.f;

        #pragma unroll
        for (int nb = 0; nb < 16; ++nb) {
            const float s0 = (nb & 8) ? -1.f : 1.f;   // n bit7 (qubit 0)
            const float s1 = (nb & 4) ? -1.f : 1.f;
            const float s2 = (nb & 2) ? -1.f : 1.f;
            const float s3 = (nb & 1) ? -1.f : 1.f;
            #pragma unroll
            for (int r = 0; r < 8; ++r) {
                float y = acc[nb][r];
                float p = y * y;
                z[r][0] += s0 * p; z[r][1] += s1 * p;
                z[r][2] += s2 * p; z[r][3] += s3 * p;
                z[r][4] = fmaf(sg4, p, z[r][4]);
                z[r][5] = fmaf(sg5, p, z[r][5]);
                z[r][6] = fmaf(sg6, p, z[r][6]);
                z[r][7] = fmaf(sg7, p, z[r][7]);
            }
        }
        // reduce over the 16 lanes of each half (covers all n)
        #pragma unroll
        for (int m = 1; m < 16; m <<= 1)
            #pragma unroll
            for (int r = 0; r < 8; ++r)
                #pragma unroll
                for (int i = 0; i < 8; ++i)
                    z[r][i] += __shfl_xor(z[r][i], m, 32);

        // rows: lanes<16 -> m=r ; lanes>=16 -> m=r+8 ; lane ln<8 writes z[r][ln]
        #pragma unroll
        for (int r = 0; r < 8; ++r) {
            float val = z[r][0];
            if (ln == 1) val = z[r][1];
            if (ln == 2) val = z[r][2];
            if (ln == 3) val = z[r][3];
            if (ln == 4) val = z[r][4];
            if (ln == 5) val = z[r][5];
            if (ln == 6) val = z[r][6];
            if (ln == 7) val = z[r][7];
            if (ln < 8) {
                long srw = sbase + r + ((lane >> 4) << 3);
                Zall[srw * 32 + v * 8 + ln] = val;
            }
        }
    }
}

// ---------------------------------------------------------------------------
// K4: gates (8->256 each) + LSTM elementwise. block = one sample.
// ---------------------------------------------------------------------------
__global__ __launch_bounds__(256) void k_gates(
    const float* __restrict__ Zall, const float* __restrict__ c_prev,
    const float* __restrict__ W_i, const float* __restrict__ b_i,
    const float* __restrict__ W_f, const float* __restrict__ b_f,
    const float* __restrict__ W_g, const float* __restrict__ b_g,
    const float* __restrict__ W_o, const float* __restrict__ b_o,
    float* __restrict__ h_out, float* __restrict__ c_out)
{
    int h = threadIdx.x;
    long s = blockIdx.x;
    const float* z = Zall + s * 32;
    float ai = b_i[h], af = b_f[h], ag = b_g[h], ao = b_o[h];
    const float* wi = W_i + h * 8;
    const float* wf = W_f + h * 8;
    const float* wg = W_g + h * 8;
    const float* wo = W_o + h * 8;
    #pragma unroll
    for (int k = 0; k < 8; ++k) {
        ai = fmaf(z[k],      wi[k], ai);
        af = fmaf(z[8 + k],  wf[k], af);
        ag = fmaf(z[16 + k], wg[k], ag);
        ao = fmaf(z[24 + k], wo[k], ao);
    }
    float i_t = 1.f / (1.f + expf(-ai));
    float f_t = 1.f / (1.f + expf(-af));
    float g_t = tanhf(ag);
    float o_t = 1.f / (1.f + expf(-ao));
    long idx = s * 256 + h;
    float c_t = f_t * c_prev[idx] + i_t * g_t;
    h_out[idx] = o_t * tanhf(c_t);
    c_out[idx] = c_t;
}

extern "C" void kernel_launch(void* const* d_in, const int* in_sizes, int n_in,
                              void* d_out, int out_size, void* d_ws, size_t ws_size,
                              hipStream_t stream) {
    const float* x_t    = (const float*)d_in[0];
    const float* h_prev = (const float*)d_in[1];
    const float* c_prev = (const float*)d_in[2];
    const float* W_in   = (const float*)d_in[3];
    const float* b_in   = (const float*)d_in[4];
    const float* ln_g   = (const float*)d_in[5];
    const float* ln_b   = (const float*)d_in[6];
    const float* w_vi   = (const float*)d_in[7];
    const float* w_vf   = (const float*)d_in[8];
    const float* w_vg   = (const float*)d_in[9];
    const float* w_vo   = (const float*)d_in[10];
    const float* W_i    = (const float*)d_in[11];
    const float* b_i    = (const float*)d_in[12];
    const float* W_f    = (const float*)d_in[13];
    const float* b_f    = (const float*)d_in[14];
    const float* W_g    = (const float*)d_in[15];
    const float* b_g    = (const float*)d_in[16];
    const float* W_o    = (const float*)d_in[17];
    const float* b_o    = (const float*)d_in[18];

    const long B = (long)in_sizes[0] / 256;     // 65536

    char* ws = (char*)d_ws;
    _Float16*     st0h = (_Float16*)ws;                                   // B*256*2
    unsigned int* UhB  = (unsigned int*)(ws + (size_t)B * 512);           // 4*32768*4
    float*        Zall = (float*)(ws + (size_t)B * 512 + 4 * 32768 * 4);  // B*32*4

    float* h_out = (float*)d_out;
    float* c_out = h_out + (size_t)B * 256;

    k_prep<<<(int)(B / 8), 256, 0, stream>>>(x_t, h_prev, W_in, b_in, ln_g, ln_b, st0h);
    k_buildU<<<8, 128, 256 * 128 * 4, stream>>>(w_vi, w_vf, w_vg, w_vo, UhB);
    k_gemm<<<(int)(B / 128), 256, 32768 * 4, stream>>>(st0h, UhB, Zall);
    k_gates<<<(int)B, 256, 0, stream>>>(Zall, c_prev, W_i, b_i, W_f, b_f,
                                        W_g, b_g, W_o, b_o, h_out, c_out);
}
